// Qwen3_5GatedDeltaNet_11020886081815
// MI455X (gfx1250) — compile-verified, matrix operand order checked by reading, not run on hardware
//
#include <hip/hip_runtime.h>
#include <hip/hip_bf16.h>

// ---------------------------------------------------------------------------
// Qwen3.5 GatedDeltaNet for MI455X (gfx1250, wave32, WMMA, async LDS DMA)
//
//  1) convert hs -> bf16; transpose weights -> bf16 [N,K]; W_b|W_a fused
//  2) WMMA bf16 GEMMs (block tile 128x64; B staged via
//     global_load_async_to_lds_b128 double-buffer; all tiles blockIdx-uniform
//     so EXEC stays all-ones around every v_wmma)
//  3) causal depthwise conv (K=4) + silu
//  4) gated delta-rule scan: 1 block/(batch,head), 128x128 state in registers
//  5) gated RMSNorm -> bf16
//  6) WMMA GEMM: out = xnorm @ W_out
// ---------------------------------------------------------------------------

typedef __attribute__((ext_vector_type(16))) __bf16 v16bf;
typedef __attribute__((ext_vector_type(8)))  __bf16 v8bf;
typedef __attribute__((ext_vector_type(8)))  float  v8f;

namespace cfg {
constexpr int B = 2, S = 2048, D = 2048;
constexpr int HK = 16, HV = 32, DK = 128, DV = 128;
constexpr int KEY_DIM = HK * DK;                  // 2048
constexpr int VALUE_DIM = HV * DV;                // 4096
constexpr int CONV_DIM = 2 * KEY_DIM + VALUE_DIM; // 8192
constexpr int KCONV = 4;
constexpr int M = B * S;                          // 4096
constexpr float EPS = 1e-6f;
}

// ---------------------------------------------------------------------------
// fp32 -> bf16 elementwise
// ---------------------------------------------------------------------------
__global__ __launch_bounds__(256) void f32_to_bf16_kernel(
    const float* __restrict__ in, __bf16* __restrict__ out, size_t n) {
  size_t i = (size_t)blockIdx.x * blockDim.x + threadIdx.x;
  if (i < n) out[i] = (__bf16)in[i];
}

// ---------------------------------------------------------------------------
// LDS-tiled transpose+convert: in fp32 [rows, cols] -> out bf16 [cols, rows]
// ---------------------------------------------------------------------------
__global__ __launch_bounds__(256) void transpose_to_bf16_kernel(
    const float* __restrict__ in, __bf16* __restrict__ out, int rows, int cols) {
  __shared__ float tile[32][33];
  const int tx = threadIdx.x & 31;
  const int ty = threadIdx.x >> 5;
  const int c0 = blockIdx.x * 32;
  const int r0 = blockIdx.y * 32;
#pragma unroll
  for (int j = 0; j < 32; j += 8) {
    int r = r0 + ty + j, c = c0 + tx;
    tile[ty + j][tx] = (r < rows && c < cols) ? in[(size_t)r * cols + c] : 0.f;
  }
  __syncthreads();
#pragma unroll
  for (int j = 0; j < 32; j += 8) {
    int c = c0 + ty + j, r = r0 + tx;
    if (c < cols && r < rows) out[(size_t)c * rows + r] = (__bf16)tile[tx][ty + j];
  }
}

// ---------------------------------------------------------------------------
// WMMA GEMM: C[M,N](fp32) = A[M,K](bf16 row-major) * Bt[N,K](bf16 row-major)
// Grid (N/64, M/128), block 128 = 4 waves. Block owns 64 N-columns (from
// blockIdx.x, uniform); wave w owns rows [blockIdx.y*128 + w*32, +32).
// B chunk (64 cols x 32 k = 4KB bf16) staged to LDS via async DMA,
// double-buffered, prefetch distance 1 chunk.
// Fragment layout per CDNA5 ISA (16-bit A 16x32 / B 32x16):
//   lanes 0-15  hold K = k0+{0..7} in elems 0..7, K = k0+{16..23} in 8..15
//   lanes 16-31 hold K = k0+{8..15} / k0+{24..31}
// ---------------------------------------------------------------------------
__global__ __launch_bounds__(128) void wmma_gemm_bf16_kernel(
    const __bf16* __restrict__ A, const __bf16* __restrict__ Bt,
    float* __restrict__ C, int Mdim, int Ndim, int Kdim) {
  __shared__ __bf16 Bsh[2][64 * 32];        // [buf][col*32 + kk], 8KB total

  const int tid = threadIdx.x;
  const int lane = tid & 31;
  const int wave = tid >> 5;                // 0..3 -> M split (no N guards)
  const int l16 = lane & 15;
  const int half = lane >> 4;               // 0 or 1
  const int m0 = blockIdx.y * 128 + wave * 32;
  const int n0 = blockIdx.x * 64;           // uniform across block

  const unsigned ldsbase = (unsigned)(uintptr_t)&Bsh[0][0];

  const __bf16* arow0 = A + (size_t)(m0 + l16) * Kdim;
  const __bf16* arow1 = A + (size_t)(m0 + 16 + l16) * Kdim;

  // async stage of one 64x32 B-chunk: 256 x 16B segments, 2 per thread
  auto stage = [&](int buf, int k0) {
#pragma unroll
    for (int s2 = 0; s2 < 2; ++s2) {
      const int seg = tid + s2 * 128;       // 0..255
      const int col = seg >> 2;
      const int ke = (seg & 3) * 8;         // k element offset within chunk
      const __bf16* src = Bt + (size_t)(n0 + col) * Kdim + k0 + ke;
      const unsigned dst =
          ldsbase + (unsigned)(buf * (64 * 32 * 2) + (col * 32 + ke) * 2);
      const unsigned long long ga = (unsigned long long)src;
      asm volatile("global_load_async_to_lds_b128 %0, %1, off"
                   :: "v"(dst), "v"(ga) : "memory");
    }
  };

  v8f acc[2][4];
#pragma unroll
  for (int r = 0; r < 2; ++r)
#pragma unroll
    for (int t = 0; t < 4; ++t)
      acc[r][t] = (v8f){0.f, 0.f, 0.f, 0.f, 0.f, 0.f, 0.f, 0.f};

  stage(0, 0);                              // preload chunk 0

  for (int k0 = 0; k0 < Kdim; k0 += 32) {
    const int buf = (k0 >> 5) & 1;

    // own DMA for current buf complete, then make all waves' writes visible
    asm volatile("s_wait_asynccnt 0x0" ::: "memory");
    __syncthreads();

    // pull all B fragments for this chunk out of LDS into registers
    v16bf bfrag[4];
#pragma unroll
    for (int t = 0; t < 4; ++t) {
      const __bf16* bp = &Bsh[buf][(t * 16 + l16) * 32 + half * 8];
      v8bf blo = *(const v8bf*)(bp);
      v8bf bhi = *(const v8bf*)(bp + 16);
#pragma unroll
      for (int j = 0; j < 8; ++j) { bfrag[t][j] = blo[j]; bfrag[t][8 + j] = bhi[j]; }
    }

    // kick off DMA of the next chunk (other buffer) to overlap with WMMA
    if (k0 + 32 < Kdim) stage(buf ^ 1, k0 + 32);

    // A fragments: two 16-row frags per wave, contiguous 16B loads
    const int kA = k0 + half * 8;
    v8bf a0lo = *(const v8bf*)(arow0 + kA);
    v8bf a0hi = *(const v8bf*)(arow0 + kA + 16);
    v8bf a1lo = *(const v8bf*)(arow1 + kA);
    v8bf a1hi = *(const v8bf*)(arow1 + kA + 16);
    if (k0 + 96 < Kdim) {
      __builtin_prefetch(arow0 + kA + 96, 0, 1);
      __builtin_prefetch(arow1 + kA + 96, 0, 1);
    }
    v16bf afrag[2];
#pragma unroll
    for (int j = 0; j < 8; ++j) {
      afrag[0][j] = a0lo[j]; afrag[0][8 + j] = a0hi[j];
      afrag[1][j] = a1lo[j]; afrag[1][8 + j] = a1hi[j];
    }

#pragma unroll
    for (int r = 0; r < 2; ++r)
#pragma unroll
      for (int t = 0; t < 4; ++t)
        acc[r][t] = __builtin_amdgcn_wmma_f32_16x16x32_bf16(
            false, afrag[r], false, bfrag[t], (short)0, acc[r][t], false, false);
  }

  // C/D layout: VGPR i -> row base + i + 8*half, col = tile + (lane&15)
#pragma unroll
  for (int r = 0; r < 2; ++r) {
    const int mb = m0 + r * 16 + half * 8;
#pragma unroll
    for (int t = 0; t < 4; ++t) {
      const int col = n0 + t * 16 + l16;
#pragma unroll
      for (int i = 0; i < 8; ++i)
        C[(size_t)(mb + i) * Ndim + col] = acc[r][t][i];
    }
  }
}

// ---------------------------------------------------------------------------
// Causal depthwise conv (K=4) + silu: x,y [B,S,CONV_DIM], w [CONV_DIM,4]
// ---------------------------------------------------------------------------
__global__ __launch_bounds__(256) void conv_silu_kernel(
    const float* __restrict__ x, const float* __restrict__ w,
    float* __restrict__ y) {
  size_t idx = (size_t)blockIdx.x * blockDim.x + threadIdx.x;
  const size_t total = (size_t)cfg::M * cfg::CONV_DIM;
  if (idx >= total) return;
  const int c = (int)(idx % cfg::CONV_DIM);
  const size_t bs = idx / cfg::CONV_DIM;
  const int s = (int)(bs % cfg::S);
  float acc = 0.f;
#pragma unroll
  for (int i = 0; i < cfg::KCONV; ++i) {
    int sp = s - (cfg::KCONV - 1) + i;
    if (sp >= 0)
      acc += w[(size_t)c * cfg::KCONV + i] *
             x[idx - (size_t)(cfg::KCONV - 1 - i) * cfg::CONV_DIM];
  }
  y[idx] = acc / (1.f + __expf(-acc));      // silu
}

// ---------------------------------------------------------------------------
// Gated delta-rule scan. One block per (batch, head): 128 threads, each owns
// one state column S[:,v] fully in registers (128 VGPRs). q/k/v broadcast
// through LDS. l2norm + beta/g activation math fused per step.
// ba: [M,64] fused projection, cols 0..31 = b, 32..63 = a.
// ---------------------------------------------------------------------------
__global__ __launch_bounds__(128) void gdn_scan_kernel(
    const float* __restrict__ qkv,          // [B,S,CONV_DIM] post conv+silu
    const float* __restrict__ ba,           // [B,S,64]
    const float* __restrict__ dt_bias,      // [HV]
    const float* __restrict__ A_log,        // [HV]
    float* __restrict__ core) {             // [B,S,HV,DV]
  const int tid = threadIdx.x;              // column v
  const int bh = blockIdx.x;                // 0..63
  const int b = bh >> 5;
  const int h = bh & 31;
  const int hk = h >> 1;                    // HV/HK = 2

  __shared__ float qsh[cfg::DK], ksh[cfg::DK], vsh[cfg::DV];

  float Sreg[cfg::DK];
#pragma unroll
  for (int k = 0; k < cfg::DK; ++k) Sreg[k] = 0.f;

  const float Aexp = __expf(A_log[h]);
  const float dtb = dt_bias[h];
  const float qmul = 0.08838834764831845f;  // DK^-0.5

  for (int s = 0; s < cfg::S; ++s) {
    const size_t row = (size_t)(b * cfg::S + s);
    const size_t ro = row * cfg::CONV_DIM;
    qsh[tid] = qkv[ro + hk * cfg::DK + tid];
    ksh[tid] = qkv[ro + cfg::KEY_DIM + hk * cfg::DK + tid];
    vsh[tid] = qkv[ro + 2 * cfg::KEY_DIM + h * cfg::DV + tid];
    __syncthreads();

    const float bval = ba[row * 64 + h];
    const float aval = ba[row * 64 + 32 + h] + dtb;
    const float beta = 1.f / (1.f + __expf(-bval));
    const float sp = (aval > 20.f) ? aval : log1pf(__expf(aval));
    const float decay = __expf(-Aexp * sp);

    float sq = 0.f, sk = 0.f;
#pragma unroll 8
    for (int i = 0; i < cfg::DK; ++i) {
      float q = qsh[i], k = ksh[i];
      sq += q * q; sk += k * k;
    }
    const float qs = rsqrtf(sq + cfg::EPS) * qmul;
    const float ks = rsqrtf(sk + cfg::EPS);

    float kv = 0.f;                         // pass 1: decay + prediction
#pragma unroll
    for (int k = 0; k < cfg::DK; ++k) {
      float sv = Sreg[k] * decay;
      Sreg[k] = sv;
      kv += (ksh[k] * ks) * sv;
    }
    const float delta = (vsh[tid] - kv) * beta;

    float o = 0.f;                          // pass 2: rank-1 update + output
#pragma unroll
    for (int k = 0; k < cfg::DK; ++k) {
      float sv = Sreg[k] + (ksh[k] * ks) * delta;
      Sreg[k] = sv;
      o += (qsh[k] * qs) * sv;
    }
    core[(row * cfg::HV + h) * cfg::DV + tid] = o;
    __syncthreads();
  }
}

// ---------------------------------------------------------------------------
// Gated RMSNorm per head: x = core * silu(z); out = x*rsqrt(mean(x^2)+eps)*w
// ---------------------------------------------------------------------------
__global__ __launch_bounds__(128) void gated_rmsnorm_kernel(
    const float* __restrict__ core, const float* __restrict__ z,
    const float* __restrict__ w, __bf16* __restrict__ out) {
  const int tid = threadIdx.x;
  const size_t row = blockIdx.x;            // 0..B*S*HV-1
  const float zz = z[row * cfg::DV + tid];
  const float x = core[row * cfg::DV + tid] * (zz / (1.f + __expf(-zz)));
  __shared__ float sh[cfg::DV];
  sh[tid] = x * x;
  __syncthreads();
  float sum = 0.f;
#pragma unroll 8
  for (int i = 0; i < cfg::DV; ++i) sum += sh[i];
  const float inv = rsqrtf(sum * (1.f / cfg::DV) + cfg::EPS);
  out[row * cfg::DV + tid] = (__bf16)(x * inv * w[tid]);
}

// ---------------------------------------------------------------------------
// Host launch
// ---------------------------------------------------------------------------
static inline void* carve(char*& p, size_t bytes) {
  void* r = (void*)p;
  p += (bytes + 255) & ~(size_t)255;
  return r;
}

extern "C" void kernel_launch(void* const* d_in, const int* in_sizes, int n_in,
                              void* d_out, int out_size, void* d_ws, size_t ws_size,
                              hipStream_t stream) {
  using namespace cfg;
  const float* hs      = (const float*)d_in[0];
  const float* W_qkv   = (const float*)d_in[1];
  const float* W_z     = (const float*)d_in[2];
  const float* W_b     = (const float*)d_in[3];
  const float* W_a     = (const float*)d_in[4];
  const float* dt_bias = (const float*)d_in[5];
  const float* A_log   = (const float*)d_in[6];
  const float* conv_w  = (const float*)d_in[7];
  const float* norm_w  = (const float*)d_in[8];
  const float* W_out   = (const float*)d_in[9];
  float* out = (float*)d_out;

  char* p = (char*)d_ws;
  __bf16* hsB    = (__bf16*)carve(p, (size_t)M * D * 2);
  __bf16* WqkvT  = (__bf16*)carve(p, (size_t)CONV_DIM * D * 2);
  __bf16* WzT    = (__bf16*)carve(p, (size_t)VALUE_DIM * D * 2);
  __bf16* WbaT   = (__bf16*)carve(p, (size_t)64 * D * 2);     // b rows 0..31, a rows 32..63
  __bf16* WoutT  = (__bf16*)carve(p, (size_t)D * VALUE_DIM * 2);
  float*  mixed  = (float*)carve(p, (size_t)M * CONV_DIM * 4);
  float*  qkvc   = (float*)carve(p, (size_t)M * CONV_DIM * 4);
  float*  zbuf   = (float*)carve(p, (size_t)M * VALUE_DIM * 4);
  float*  babuf  = (float*)carve(p, (size_t)M * 64 * 4);
  __bf16* xnorm  = (__bf16*)carve(p, (size_t)M * VALUE_DIM * 2);
  float*  core   = mixed;  // reuse: mixed is dead after conv_silu_kernel

  // 1) conversions / transposes
  {
    size_t n = (size_t)M * D;
    f32_to_bf16_kernel<<<dim3((unsigned)((n + 255) / 256)), 256, 0, stream>>>(hs, hsB, n);
  }
  transpose_to_bf16_kernel<<<dim3(CONV_DIM / 32, D / 32), 256, 0, stream>>>(W_qkv, WqkvT, D, CONV_DIM);
  transpose_to_bf16_kernel<<<dim3(VALUE_DIM / 32, D / 32), 256, 0, stream>>>(W_z, WzT, D, VALUE_DIM);
  transpose_to_bf16_kernel<<<dim3(1, D / 32), 256, 0, stream>>>(W_b, WbaT, D, HV);
  transpose_to_bf16_kernel<<<dim3(1, D / 32), 256, 0, stream>>>(W_a, WbaT + (size_t)HV * D, D, HV);
  transpose_to_bf16_kernel<<<dim3(D / 32, VALUE_DIM / 32), 256, 0, stream>>>(W_out, WoutT, VALUE_DIM, D);

  // 2) projection GEMMs (WMMA bf16, fp32 accumulate); N % 64 == 0 everywhere
  wmma_gemm_bf16_kernel<<<dim3(CONV_DIM / 64, M / 128), 128, 0, stream>>>(
      hsB, WqkvT, mixed, M, CONV_DIM, D);
  wmma_gemm_bf16_kernel<<<dim3(VALUE_DIM / 64, M / 128), 128, 0, stream>>>(
      hsB, WzT, zbuf, M, VALUE_DIM, D);
  wmma_gemm_bf16_kernel<<<dim3(1, M / 128), 128, 0, stream>>>(
      hsB, WbaT, babuf, M, 64, D);

  // 3) causal depthwise conv + silu
  {
    size_t n = (size_t)M * CONV_DIM;
    conv_silu_kernel<<<dim3((unsigned)((n + 255) / 256)), 256, 0, stream>>>(mixed, conv_w, qkvc);
  }

  // 4) gated delta-rule scan (core overwrites mixed — conv already consumed it)
  gdn_scan_kernel<<<dim3(B * HV), 128, 0, stream>>>(qkvc, babuf, dt_bias, A_log, core);

  // 5) gated RMSNorm -> bf16
  gated_rmsnorm_kernel<<<dim3(M * HV), 128, 0, stream>>>(core, zbuf, norm_w, xnorm);

  // 6) output projection
  wmma_gemm_bf16_kernel<<<dim3(D / 64, M / 128), 128, 0, stream>>>(
      xnorm, WoutT, out, M, D, VALUE_DIM);
}
